// StackedLSTM_76063870812626
// MI455X (gfx1250) — matrix operand order repaired and checked by reading, not compiled, not run on hardware
//
#include <hip/hip_runtime.h>
#include <math.h>

#define DEVINL __device__ __forceinline__

typedef __attribute__((ext_vector_type(16))) __bf16 bf16x16;
typedef __attribute__((ext_vector_type(8)))  float  f32x8;

constexpr int Bc  = 32;     // batch
constexpr int Tc  = 256;    // timesteps
constexpr int DIc = 512;    // input dim
constexpr int Hc  = 1024;   // hidden
constexpr int G4  = 4 * Hc; // 4096 gate cols (layer 0)
constexpr int G5  = 5 * Hc; // 5120 gate cols (ca layers)

union FragU { bf16x16 v; uint4 q[2]; };

DEVINL unsigned short f32_to_bf16(float f) {
  unsigned u = __float_as_uint(f);
  u += 0x7FFFu + ((u >> 16) & 1u);           // round-to-nearest-even
  return (unsigned short)(u >> 16);
}

// A fragment: 16x32 bf16 tile from row-major source (ld elements per row).
// ISA layout: lane L holds row m=L&15; K chunks [8*(L>>4),+8) and [16+8*(L>>4),+8).
DEVINL bf16x16 load_frag_a(const unsigned short* p0, int ld) {
  int lane = threadIdx.x & 31;
  const unsigned short* p = p0 + (size_t)(lane & 15) * ld + ((lane >> 4) << 3);
  FragU f;
  f.q[0] = *(const uint4*)(p);
  f.q[1] = *(const uint4*)(p + 16);
  return f.v;
}

// B fragment: 32x16 (KxN) tile. W stored (N,K) row-major == Bᵀ, pointer at (n0,k0).
// ISA layout: lane L holds column n=L&15; K chunks [8*(L>>4),+8) and [16+8*(L>>4),+8).
DEVINL bf16x16 load_frag_b(const unsigned short* w0, int ld) {
  int lane = threadIdx.x & 31;
  const unsigned short* p = w0 + (size_t)(lane & 15) * ld + ((lane >> 4) << 3);
  FragU f;
  f.q[0] = *(const uint4*)(p);
  f.q[1] = *(const uint4*)(p + 16);
  return f.v;
}

DEVINL f32x8 wmma_bf16(bf16x16 a, bf16x16 b, f32x8 c) {
  return __builtin_amdgcn_wmma_f32_16x16x32_bf16(false, a, false, b, (short)0, c,
                                                 false, false);
}

DEVINL float sigmoidf_(float x) { return 1.f / (1.f + __expf(-x)); }

// ---------------------------------------------------------------- pack kernels
__global__ __launch_bounds__(256) void k_pack_x(const float* __restrict__ x,
                                                unsigned short* __restrict__ out) {
  int idx = blockIdx.x * 256 + threadIdx.x;          // over B*T*DI
  if (idx >= Bc * Tc * DIc) return;
  int d  = idx % DIc;
  int bt = idx / DIc;
  int t  = bt % Tc;
  int b  = bt / Tc;
  out[((size_t)t * Bc + b) * DIc + d] = f32_to_bf16(x[idx]);  // (T*B, DI)
}

__global__ __launch_bounds__(256) void k_cvt_bf16(const float* __restrict__ in,
                                                  unsigned short* __restrict__ out,
                                                  int n) {
  int idx = blockIdx.x * 256 + threadIdx.x;
  if (idx < n) out[idx] = f32_to_bf16(in[idx]);
}

// -------------------------------------------------------- batched WMMA GEMM
// C(M,N) = A(M,K)bf16 * W(N,K)bf16ᵀ + bias0[n] + bias1[n]
// block = 8 waves; wave w owns a 16x64 strip. Software-pipelined (fragments for
// iteration k+1 loaded before the WMMAs of iteration k -> partial
// s_wait_loadcnt) and unrolled x2 so the double-buffer register rotation is
// absorbed into the unroll instead of v_mov chains.
__global__ __launch_bounds__(256) void k_gemm_bias(
    const unsigned short* __restrict__ A, int lda,
    const unsigned short* __restrict__ W, int ldw,
    const float* __restrict__ bias0, const float* __restrict__ bias1,
    float* __restrict__ C, int ldc, int K) {
  int wave = threadIdx.x >> 5;
  int lane = threadIdx.x & 31;
  int m0 = blockIdx.x * 16;
  int n0 = (blockIdx.y * 8 + wave) * 64;
  f32x8 acc[4] = {};
  const unsigned short* arow = A + (size_t)m0 * lda;
  const unsigned short* wrow[4];
#pragma unroll
  for (int j = 0; j < 4; ++j) wrow[j] = W + (size_t)(n0 + j * 16) * ldw;

  bf16x16 a = load_frag_a(arow, lda);
  bf16x16 b[4];
#pragma unroll
  for (int j = 0; j < 4; ++j) b[j] = load_frag_b(wrow[j], ldw);

#pragma unroll 2
  for (int k = 32; k < K; k += 32) {
    bf16x16 a2 = load_frag_a(arow + k, lda);
    __builtin_prefetch(arow + k + 32, 0, 1);          // global_prefetch_b8
    bf16x16 b2[4];
#pragma unroll
    for (int j = 0; j < 4; ++j) {
      b2[j] = load_frag_b(wrow[j] + k, ldw);
      __builtin_prefetch(wrow[j] + k + 32, 0, 1);
    }
#pragma unroll
    for (int j = 0; j < 4; ++j) acc[j] = wmma_bf16(a, b[j], acc[j]);
    a = a2;
#pragma unroll
    for (int j = 0; j < 4; ++j) b[j] = b2[j];
  }
#pragma unroll
  for (int j = 0; j < 4; ++j) acc[j] = wmma_bf16(a, b[j], acc[j]);

  int n  = lane & 15;
  int mh = (lane >> 4) << 3;                          // C layout: m = mh + r
#pragma unroll
  for (int j = 0; j < 4; ++j) {
    int col  = n0 + j * 16 + n;
    float bv = (bias0 ? bias0[col] : 0.f) + (bias1 ? bias1[col] : 0.f);
#pragma unroll
    for (int r = 0; r < 8; ++r)
      C[(size_t)(m0 + mh + r) * ldc + col] = acc[j][r] + bv;
  }
}

// ------------------------------------------------- fused layer-0 LSTM step
// gates = xp_t + h_prev @ Whhᵀ ; block owns 64 h-columns, 8 waves = 2 mTiles x 4 gates
__global__ __launch_bounds__(256) void k_lstm0_step(
    const float* __restrict__ xp_t,            // (B, 4H)
    const unsigned short* __restrict__ Whh,    // (4H, H) bf16
    const unsigned short* __restrict__ hprev,  // (B, H) bf16 or null (t==0)
    const float* __restrict__ cprev,           // (B, H) or null
    unsigned short* __restrict__ hbf_out,      // (B, H)
    float* __restrict__ c_out)                 // (B, H)
{
  __shared__ float gbuf[4][Bc][64];
  int wave = threadIdx.x >> 5;
  int lane = threadIdx.x & 31;
  int mT = wave & 1;
  int g  = wave >> 1;
  int j0 = blockIdx.x * 64;
  f32x8 acc[4] = {};
  if (hprev) {
    const unsigned short* arow = hprev + (size_t)mT * 16 * Hc;
    const unsigned short* wrow[4];
#pragma unroll
    for (int j = 0; j < 4; ++j)
      wrow[j] = Whh + (size_t)(g * Hc + j0 + j * 16) * Hc;

    bf16x16 a = load_frag_a(arow, Hc);
    bf16x16 b[4];
#pragma unroll
    for (int j = 0; j < 4; ++j) b[j] = load_frag_b(wrow[j], Hc);

#pragma unroll 2
    for (int k = 32; k < Hc; k += 32) {
      bf16x16 a2 = load_frag_a(arow + k, Hc);
      bf16x16 b2[4];
#pragma unroll
      for (int j = 0; j < 4; ++j) {
        b2[j] = load_frag_b(wrow[j] + k, Hc);
        __builtin_prefetch(wrow[j] + k + 32, 0, 1);
      }
#pragma unroll
      for (int j = 0; j < 4; ++j) acc[j] = wmma_bf16(a, b[j], acc[j]);
      a = a2;
#pragma unroll
      for (int j = 0; j < 4; ++j) b[j] = b2[j];
    }
#pragma unroll
    for (int j = 0; j < 4; ++j) acc[j] = wmma_bf16(a, b[j], acc[j]);
  }
  int n  = lane & 15;
  int mh = mT * 16 + ((lane >> 4) << 3);
#pragma unroll
  for (int j = 0; j < 4; ++j)
#pragma unroll
    for (int r = 0; r < 8; ++r) {
      int row = mh + r;                                // batch 0..31
      int col = g * Hc + j0 + j * 16 + n;
      gbuf[g][row][j * 16 + n] = acc[j][r] + xp_t[(size_t)row * G4 + col];
    }
  __syncthreads();
  for (int e = threadIdx.x; e < Bc * 64; e += 256) {
    int b  = e >> 6;
    int jj = e & 63;
    int j  = j0 + jj;
    float iv = gbuf[0][b][jj];
    float fv = gbuf[1][b][jj];
    float gv = gbuf[2][b][jj];
    float ov = gbuf[3][b][jj];
    float cp = cprev ? cprev[(size_t)b * Hc + j] : 0.f;
    float c  = sigmoidf_(fv) * cp + sigmoidf_(iv) * tanhf(gv);
    float h  = sigmoidf_(ov) * tanhf(c);
    c_out[(size_t)b * Hc + j]   = c;
    hbf_out[(size_t)b * Hc + j] = f32_to_bf16(h);
  }
}

// ------------------------------------------------- fused cell-attached step
// gates = pre_t + h_prev @ Wprevᵀ (5 gates).
// 8 waves = 2 mTiles x 4 jTiles; each wave holds 5 gate accumulators so the A
// fragment is loaded once per k-step and amortized over 5 WMMAs.
__global__ __launch_bounds__(256) void k_ca_step(
    const float* __restrict__ pre_t,            // (B, 5H)
    const unsigned short* __restrict__ Wprev,   // (5H, ldw), col offset applied
    int ldw,
    const unsigned short* __restrict__ hprev,   // (B, H) bf16 or null
    const float* __restrict__ cprev, int cprev_sb,
    const float* __restrict__ clow_t,           // (B, H) stride H
    unsigned short* __restrict__ hbf_out,       // (B, H)
    float* __restrict__ c_out, int c_sb,
    float* __restrict__ hf_out, int hf_sb)      // null for intermediate layers
{
  __shared__ float gbuf[5][Bc][64];
  int wave = threadIdx.x >> 5;
  int lane = threadIdx.x & 31;
  int mT = wave >> 2;
  int jt = wave & 3;
  int j0 = blockIdx.x * 64;
  int n  = lane & 15;
  f32x8 acc[5] = {};
  if (hprev) {
    const unsigned short* arow = hprev + (size_t)mT * 16 * Hc;
    const unsigned short* wrow[5];
#pragma unroll
    for (int g = 0; g < 5; ++g)
      wrow[g] = Wprev + (size_t)(g * Hc + j0 + jt * 16) * ldw;

    bf16x16 a = load_frag_a(arow, Hc);
    bf16x16 b[5];
#pragma unroll
    for (int g = 0; g < 5; ++g) b[g] = load_frag_b(wrow[g], ldw);

#pragma unroll 2
    for (int k = 32; k < Hc; k += 32) {
      bf16x16 a2 = load_frag_a(arow + k, Hc);
      bf16x16 b2[5];
#pragma unroll
      for (int g = 0; g < 5; ++g) {
        b2[g] = load_frag_b(wrow[g] + k, ldw);
        __builtin_prefetch(wrow[g] + k + 32, 0, 1);
      }
#pragma unroll
      for (int g = 0; g < 5; ++g) acc[g] = wmma_bf16(a, b[g], acc[g]);
      a = a2;
#pragma unroll
      for (int g = 0; g < 5; ++g) b[g] = b2[g];
    }
#pragma unroll
    for (int g = 0; g < 5; ++g) acc[g] = wmma_bf16(a, b[g], acc[g]);
  }
  int mh = mT * 16 + ((lane >> 4) << 3);
#pragma unroll
  for (int g = 0; g < 5; ++g)
#pragma unroll
    for (int r = 0; r < 8; ++r) {
      int row = mh + r;
      int col = g * Hc + j0 + jt * 16 + n;
      gbuf[g][row][jt * 16 + n] = acc[g][r] + pre_t[(size_t)row * G5 + col];
    }
  __syncthreads();
  for (int e = threadIdx.x; e < Bc * 64; e += 256) {
    int b  = e >> 6;
    int jj = e & 63;
    int j  = j0 + jj;
    float iv  = gbuf[0][b][jj];
    float fpv = gbuf[1][b][jj];
    float flv = gbuf[2][b][jj];
    float uv  = gbuf[3][b][jj];
    float ov  = gbuf[4][b][jj];
    float cp = cprev ? cprev[(size_t)b * cprev_sb + j] : 0.f;
    float cl = clow_t[(size_t)b * Hc + j];
    float c  = cp * sigmoidf_(fpv + 1.f) + cl * sigmoidf_(flv + 1.f) +
               tanhf(uv) * sigmoidf_(iv);
    float h  = sigmoidf_(ov) * tanhf(c);
    c_out[(size_t)b * c_sb + j] = c;
    hbf_out[(size_t)b * Hc + j] = f32_to_bf16(h);
    if (hf_out) hf_out[(size_t)b * hf_sb + j] = h;
  }
}

// ---------------------------------------------------------------- host driver
extern "C" void kernel_launch(void* const* d_in, const int* in_sizes, int n_in,
                              void* d_out, int out_size, void* d_ws, size_t ws_size,
                              hipStream_t stream) {
  (void)in_sizes; (void)n_in; (void)out_size; (void)ws_size;
  const float* x     = (const float*)d_in[0];
  const float* W_ih0 = (const float*)d_in[1];
  const float* W_hh0 = (const float*)d_in[2];
  const float* b_ih0 = (const float*)d_in[3];
  const float* b_hh0 = (const float*)d_in[4];
  const float* W_ca  = (const float*)d_in[5];
  const float* b_ca  = (const float*)d_in[6];

  // Scratch layout (~353 MB total), 256B-aligned carve-out from d_ws.
  char* p = (char*)d_ws;
  auto take = [&](size_t bytes) -> char* {
    char* r = p;
    p += (bytes + 255) & ~(size_t)255;
    return r;
  };
  unsigned short* Xbf  = (unsigned short*)take((size_t)Tc * Bc * DIc * 2);
  unsigned short* Wih  = (unsigned short*)take((size_t)G4 * DIc * 2);
  unsigned short* Whh  = (unsigned short*)take((size_t)G4 * Hc * 2);
  unsigned short* Wca  = (unsigned short*)take((size_t)3 * G5 * 2 * Hc * 2);
  float*          preb = (float*)take((size_t)Tc * Bc * G5 * 4);  // shared xp/pre
  unsigned short* hbfA = (unsigned short*)take((size_t)Tc * Bc * Hc * 2);
  unsigned short* hbfB = (unsigned short*)take((size_t)Tc * Bc * Hc * 2);
  float*          csA  = (float*)take((size_t)Tc * Bc * Hc * 4);
  float*          csB  = (float*)take((size_t)Tc * Bc * Hc * 4);

  // fp32 -> bf16 packing
  {
    int nx = Bc * Tc * DIc;
    k_pack_x<<<(nx + 255) / 256, 256, 0, stream>>>(x, Xbf);
    int n1 = G4 * DIc;
    k_cvt_bf16<<<(n1 + 255) / 256, 256, 0, stream>>>(W_ih0, Wih, n1);
    int n2 = G4 * Hc;
    k_cvt_bf16<<<(n2 + 255) / 256, 256, 0, stream>>>(W_hh0, Whh, n2);
    int n3 = 3 * G5 * 2 * Hc;
    k_cvt_bf16<<<(n3 + 255) / 256, 256, 0, stream>>>(W_ca, Wca, n3);
  }

  const int M = Tc * Bc;  // 8192 rows (t*B + b)

  // Layer 0: input projection, then fused recurrent steps.
  k_gemm_bias<<<dim3(M / 16, G4 / 512), 256, 0, stream>>>(
      Xbf, DIc, Wih, DIc, b_ih0, b_hh0, preb, G4, DIc);
  for (int t = 0; t < Tc; ++t) {
    k_lstm0_step<<<Hc / 64, 256, 0, stream>>>(
        preb + (size_t)t * Bc * G4, Whh,
        t ? hbfA + (size_t)(t - 1) * Bc * Hc : nullptr,
        t ? csA + (size_t)(t - 1) * Bc * Hc : nullptr,
        hbfA + (size_t)t * Bc * Hc,
        csA + (size_t)t * Bc * Hc);
  }

  float* outh = (float*)d_out;                    // hs (B,T,H)
  float* outc = outh + (size_t)Bc * Tc * Hc;      // cs (B,T,H)

  // Layers 1..3 (cell-attached).
  for (int l = 1; l < 4; ++l) {
    unsigned short* hlow = (l == 2) ? hbfB : hbfA;
    float*          clow = (l == 2) ? csB : csA;
    unsigned short* hcur = (l == 2) ? hbfA : hbfB;
    float*          ccur = (l == 2) ? csA : csB;
    const unsigned short* Wl = Wca + (size_t)(l - 1) * G5 * 2 * Hc;
    const float*          bl = b_ca + (size_t)(l - 1) * G5;

    // pre = h_low @ W_lowᵀ + b   (W_low = Wl[:, :H], row stride 2H)
    k_gemm_bias<<<dim3(M / 16, G5 / 512), 256, 0, stream>>>(
        hlow, Hc, Wl, 2 * Hc, bl, nullptr, preb, G5, Hc);

    for (int t = 0; t < Tc; ++t) {
      if (l < 3) {
        k_ca_step<<<Hc / 64, 256, 0, stream>>>(
            preb + (size_t)t * Bc * G5, Wl + Hc, 2 * Hc,
            t ? hcur + (size_t)(t - 1) * Bc * Hc : nullptr,
            t ? ccur + (size_t)(t - 1) * Bc * Hc : nullptr, Hc,
            clow + (size_t)t * Bc * Hc,
            hcur + (size_t)t * Bc * Hc,
            ccur + (size_t)t * Bc * Hc, Hc,
            nullptr, 0);
      } else {
        // Final layer writes fp32 h,c straight to d_out in (B,T,H) layout and
        // reads its own c_prev back from d_out.
        k_ca_step<<<Hc / 64, 256, 0, stream>>>(
            preb + (size_t)t * Bc * G5, Wl + Hc, 2 * Hc,
            t ? hcur + (size_t)(t - 1) * Bc * Hc : nullptr,
            t ? outc + (size_t)(t - 1) * Hc : nullptr, Tc * Hc,
            clow + (size_t)t * Bc * Hc,
            hcur + (size_t)t * Bc * Hc,
            outc + (size_t)t * Hc, Tc * Hc,
            outh + (size_t)t * Hc, Tc * Hc);
      }
    }
  }
}
